// Encoder_26371099197528
// MI455X (gfx1250) — matrix operand order repaired and checked by reading, not compiled, not run on hardware
//
#include <hip/hip_runtime.h>
#include <cstdint>
#include <cstddef>

// ---------------- problem constants ----------------
#define BATCH 128
#define LSEQ  128
#define TSEQ  8
#define EDIM  300
#define EPAD  320          // 10 * 32 K-chunks for bf16 WMMA
#define HDIM  256          // 8  * 32 K-chunks
#define G4    1024         // 4*H gate width
#define NOUT  3

typedef __attribute__((ext_vector_type(8)))  float  v8f;
typedef __attribute__((ext_vector_type(8)))  __bf16 v8bf;
typedef __attribute__((ext_vector_type(16))) __bf16 v16bf;

static __device__ inline v8f v8f_zero() {
    v8f z;
#pragma unroll
    for (int i = 0; i < 8; ++i) z[i] = 0.0f;
    return z;
}

static __device__ inline v16bf cat16(v8bf lo, v8bf hi) {
    return __builtin_shufflevector(lo, hi, 0,1,2,3,4,5,6,7,8,9,10,11,12,13,14,15);
}

static __device__ inline v8f wmma_bf16(v16bf a, v16bf b, v8f c) {
#if defined(__AMDGCN__)
    // D = A(16x32 bf16) * B(32x16 bf16) + C(16x16 f32)
    return __builtin_amdgcn_wmma_f32_16x16x32_bf16(false, a, false, b, (short)0, c, false, false);
#else
    (void)a; (void)b;
    return c;
#endif
}

static __device__ inline float sigmoidf_(float x) {
    return 1.0f / (1.0f + __expf(-x));
}

// ---------------- K1: embedding gather + cast to bf16, pad E 300->320 ----------------
__global__ void embed_kernel(const int* __restrict__ idx, const float* __restrict__ emb,
                             __bf16* __restrict__ out, int nrows) {
    int r = blockIdx.x;
    int k = threadIdx.x;                 // blockDim.x == EPAD
    if (r >= nrows) return;
    int id = idx[r];
    float v = 0.0f;
    if (id != 0 && k < EDIM)             // padding_idx = 0 -> zero row
        v = emb[(size_t)id * EDIM + k];
    out[(size_t)r * EPAD + k] = (__bf16)v;
}

// ---------------- K2: cast f32 weights -> bf16, optional column pad ----------------
__global__ void castw_kernel(const float* __restrict__ src, __bf16* __restrict__ dst,
                             int cols, int colspad) {
    int r = blockIdx.x;                  // 1024 rows
    int k = threadIdx.x;                 // blockDim.x == colspad
    float v = (k < cols) ? src[(size_t)r * cols + k] : 0.0f;
    dst[(size_t)r * colspad + k] = (__bf16)v;
}

// ---------------- K3: xW = X @ Wih^T + bih + bhh   (bf16 WMMA, f32 out) ----------------
// X: [M, EPAD] bf16 row-major.  Wb: [G4, EPAD] bf16 row-major (B-frag = rows of Wih).
// grid = (M/16, 2), block = 256 (8 waves); wave w owns 4 N-tiles.
__global__ void xw_gemm_kernel(const __bf16* __restrict__ X, const __bf16* __restrict__ Wb,
                               const float* __restrict__ bih, const float* __restrict__ bhh,
                               float* __restrict__ out, int M) {
    const int tid  = threadIdx.x;
    const int w    = tid >> 5;
    const int lane = tid & 31;
    const int ln   = lane & 15;
    const bool hi  = (lane >= 16);
    const int mt   = blockIdx.x;

    const int arow = mt * 16 + ln;
    const __bf16* Aptr = X + (size_t)arow * EPAD;

    v8f acc[4];
#pragma unroll
    for (int q = 0; q < 4; ++q) acc[q] = v8f_zero();

#pragma unroll 2
    for (int kc = 0; kc < EPAD / 32; ++kc) {
        // A fragment (contiguous K layout: lane<16 -> K{0..15}, lane>=16 -> K{16..31})
        const int ab = kc * 32 + (hi ? 16 : 0);
        v8bf alo = *(const v8bf*)(Aptr + ab);
        v8bf ahi = *(const v8bf*)(Aptr + ab + 8);
        v16bf af = cat16(alo, ahi);
#pragma unroll
        for (int q = 0; q < 4; ++q) {
            const int nt = blockIdx.y * 32 + w * 4 + q;
            const int n  = nt * 16 + ln;                // B column = Wih row n (contiguous)
            const __bf16* bp = Wb + (size_t)n * EPAD + kc * 32 + (hi ? 16 : 0);
            v8bf blo = *(const v8bf*)bp;
            v8bf bhi = *(const v8bf*)(bp + 8);
            acc[q] = wmma_bf16(af, cat16(blo, bhi), acc[q]);
        }
    }

#pragma unroll
    for (int q = 0; q < 4; ++q) {
        const int nt  = blockIdx.y * 32 + w * 4 + q;
        const int col = nt * 16 + ln;
        const float bias = bih[col] + bhh[col];
#pragma unroll
        for (int j = 0; j < 8; ++j) {
            const int row = mt * 16 + j + (hi ? 8 : 0);
            out[(size_t)row * G4 + col] = acc[q][j] + bias;
        }
    }
}

// ---------------- K4: persistent LSTM recurrence, 1 block per direction ----------------
// block = 1024 threads = 32 waves; wave w: m-tile = w&7, j-group = w>>3 (4 j-tiles each).
// Each wave computes gate columns {j, j+256, j+512, j+768} so c stays in registers.
// h double-buffered in LDS as bf16 (2*128*256*2 = 128 KB dynamic LDS).
// A fragments (h rows) for all 8 K-chunks are hoisted into registers once per step,
// before any LDS stores, so each is loaded once and reused by all 16 output tiles.
__global__ void lstm_kernel(const __bf16* __restrict__ Whh_all,
                            const float* __restrict__ xW_s, const float* __restrict__ xW_t,
                            float* __restrict__ hseq_s, float* __restrict__ hseq_t) {
    extern __shared__ __bf16 hlds[];     // [2][128][256]

    const int l   = blockIdx.x;          // 0=sf 1=sb 2=tf 3=tb
    const int T   = (l < 2) ? LSEQ : TSEQ;
    const bool fwd = ((l & 1) == 0);
    const int dircol = (l & 1) * HDIM;
    const float* xW   = (l < 2) ? (xW_s + (size_t)l * BATCH * LSEQ * G4)
                                : (xW_t + (size_t)(l - 2) * BATCH * TSEQ * G4);
    float* hseq       = (l < 2) ? hseq_s : hseq_t;
    const __bf16* Whh = Whh_all + (size_t)l * G4 * HDIM;

    const int tid  = threadIdx.x;
    const int w    = tid >> 5;
    const int lane = tid & 31;
    const int ln   = lane & 15;
    const bool hi  = (lane >= 16);
    const int mt   = w & 7;
    const int jg   = w >> 3;

    // zero h(t=-1) buffer 0
    for (int i = tid; i < BATCH * HDIM; i += 1024) hlds[i] = (__bf16)0.0f;
    __syncthreads();

    float c[4][8];
#pragma unroll
    for (int q = 0; q < 4; ++q)
#pragma unroll
        for (int j = 0; j < 8; ++j) c[q][j] = 0.0f;

    const int arow = mt * 16 + ln;

    for (int s = 0; s < T; ++s) {
        const int tt = fwd ? s : (T - 1 - s);
        const int tn = fwd ? (s + 1) : (T - 2 - s);      // next step's time index
        const bool has_next = (s + 1 < T);
        const __bf16* hr = hlds + (size_t)(s & 1) * BATCH * HDIM;
        __bf16*       hw = hlds + (size_t)((s & 1) ^ 1) * BATCH * HDIM;

        // hoist all 8 A fragments (this wave's 16 h-rows) out of the tile loops
        v16bf afr[HDIM / 32];
#pragma unroll
        for (int kc = 0; kc < HDIM / 32; ++kc) {
            const int ab = kc * 32 + (hi ? 16 : 0);
            v8bf alo = *(const v8bf*)(hr + (size_t)arow * HDIM + ab);
            v8bf ahi = *(const v8bf*)(hr + (size_t)arow * HDIM + ab + 8);
            afr[kc] = cat16(alo, ahi);
        }

#pragma unroll
        for (int q = 0; q < 4; ++q) {
            const int jt = jg * 4 + q;
            v8f acc[4];
#pragma unroll
            for (int g = 0; g < 4; ++g) acc[g] = v8f_zero();

#pragma unroll
            for (int kc = 0; kc < HDIM / 32; ++kc) {
#pragma unroll
                for (int g = 0; g < 4; ++g) {
                    const int n = g * HDIM + jt * 16 + ln;      // gate column = Whh row n
                    const __bf16* bp = Whh + (size_t)n * HDIM + kc * 32 + (hi ? 16 : 0);
                    v8bf blo = *(const v8bf*)bp;
                    v8bf bhi = *(const v8bf*)(bp + 8);
                    acc[g] = wmma_bf16(afr[kc], cat16(blo, bhi), acc[g]);
                }
            }

            // gate nonlinearity + cell update; c stays in registers for this wave
#pragma unroll
            for (int j = 0; j < 8; ++j) {
                const int row = mt * 16 + j + (hi ? 8 : 0);      // batch index
                const int col = jt * 16 + ln;                    // hidden index 0..255
                const size_t xr = ((size_t)row * T + tt) * G4;
                const float gi = acc[0][j] + xW[xr + 0 * HDIM + col];
                const float gf = acc[1][j] + xW[xr + 1 * HDIM + col];
                const float gg = acc[2][j] + xW[xr + 2 * HDIM + col];
                const float go = acc[3][j] + xW[xr + 3 * HDIM + col];
                if (has_next) {
                    // prefetch next timestep's gate pre-activations (global_prefetch_b8)
                    const size_t xn = ((size_t)row * T + tn) * G4;
                    __builtin_prefetch(&xW[xn + 0 * HDIM + col], 0, 1);
                    __builtin_prefetch(&xW[xn + 1 * HDIM + col], 0, 1);
                    __builtin_prefetch(&xW[xn + 2 * HDIM + col], 0, 1);
                    __builtin_prefetch(&xW[xn + 3 * HDIM + col], 0, 1);
                }
                const float cn = sigmoidf_(gf) * c[q][j] + sigmoidf_(gi) * tanhf(gg);
                c[q][j] = cn;
                const float hn = sigmoidf_(go) * tanhf(cn);
                hw[(size_t)row * HDIM + col] = (__bf16)hn;
                hseq[((size_t)row * T + tt) * (2 * HDIM) + dircol + col] = hn;
            }
        }
        __syncthreads();
    }
}

// ---------------- K5: attention + classifier head (tiny, VALU) ----------------
__global__ void attn_kernel(const float* __restrict__ hseq_s, const float* __restrict__ hseq_t,
                            const float* __restrict__ Wout, const float* __restrict__ bout,
                            float* __restrict__ out) {
    __shared__ float A[LSEQ][TSEQ];
    __shared__ float mc[TSEQ], dc[TSEQ], rv[TSEQ];
    __shared__ float mr[LSEQ], dr[LSEQ], attn[LSEQ];
    __shared__ float score[2 * HDIM];
    const int b = blockIdx.x;
    const int tid = threadIdx.x;
    const float* sen = hseq_s + (size_t)b * LSEQ * (2 * HDIM);
    const float* tgt = hseq_t + (size_t)b * TSEQ * (2 * HDIM);

    // A[s][t] = <sen_h[b,s,:], tgt_h[b,t,:]>   (float4-vectorized dots)
    for (int p = tid; p < LSEQ * TSEQ; p += 256) {
        const int s = p >> 3, t = p & 7;
        const float4* sp = (const float4*)(sen + (size_t)s * (2 * HDIM));
        const float4* tp = (const float4*)(tgt + (size_t)t * (2 * HDIM));
        float acc = 0.0f;
        for (int h = 0; h < (2 * HDIM) / 4; ++h) {
            float4 a4 = sp[h], b4 = tp[h];
            acc += a4.x * b4.x + a4.y * b4.y + a4.z * b4.z + a4.w * b4.w;
        }
        A[s][t] = acc;
    }
    __syncthreads();

    if (tid < LSEQ) {                     // row softmax stats (over t)
        const int s = tid;
        float m = -3.4e38f;
        for (int t = 0; t < TSEQ; ++t) m = fmaxf(m, A[s][t]);
        float d = 0.0f;
        for (int t = 0; t < TSEQ; ++t) d += __expf(A[s][t] - m);
        mr[s] = m; dr[s] = d;
    }
    if (tid >= 128 && tid < 128 + TSEQ) { // col softmax stats (over s)
        const int t = tid - 128;
        float m = -3.4e38f;
        for (int s = 0; s < LSEQ; ++s) m = fmaxf(m, A[s][t]);
        float d = 0.0f;
        for (int s = 0; s < LSEQ; ++s) d += __expf(A[s][t] - m);
        mc[t] = m; dc[t] = d;
    }
    __syncthreads();

    if (tid < TSEQ) {                     // row_vec[t] = mean_s row[s][t]
        const int t = tid;
        float acc = 0.0f;
        for (int s = 0; s < LSEQ; ++s) acc += __expf(A[s][t] - mr[s]) / dr[s];
        rv[t] = acc * (1.0f / (float)LSEQ);
    }
    __syncthreads();

    if (tid < LSEQ) {                     // attn[s] = sum_t col[s][t] * row_vec[t]
        const int s = tid;
        float acc = 0.0f;
        for (int t = 0; t < TSEQ; ++t) acc += (__expf(A[s][t] - mc[t]) / dc[t]) * rv[t];
        attn[s] = acc;
    }
    __syncthreads();

    for (int hh = tid; hh < 2 * HDIM; hh += 256) {   // score = attn^T @ sen_h
        float acc = 0.0f;
        for (int s = 0; s < LSEQ; ++s) acc += attn[s] * sen[(size_t)s * (2 * HDIM) + hh];
        score[hh] = acc;
    }
    __syncthreads();

    if (tid == 0) {                       // logits + softmax
        float lg[NOUT];
        for (int o = 0; o < NOUT; ++o) {
            float acc = bout[o];
            for (int h = 0; h < 2 * HDIM; ++h) acc += score[h] * Wout[(size_t)o * (2 * HDIM) + h];
            lg[o] = acc;
        }
        float m = fmaxf(lg[0], fmaxf(lg[1], lg[2]));
        float e0 = __expf(lg[0] - m), e1 = __expf(lg[1] - m), e2 = __expf(lg[2] - m);
        float d = e0 + e1 + e2;
        out[b * NOUT + 0] = e0 / d;
        out[b * NOUT + 1] = e1 / d;
        out[b * NOUT + 2] = e2 / d;
    }
}

// ---------------- host-side orchestration ----------------
extern "C" void kernel_launch(void* const* d_in, const int* in_sizes, int n_in,
                              void* d_out, int out_size, void* d_ws, size_t ws_size,
                              hipStream_t stream) {
    (void)in_sizes; (void)n_in; (void)out_size; (void)ws_size;

    const int*   sent_idx = (const int*)d_in[0];
    const int*   tgt_idx  = (const int*)d_in[1];
    const float* embW     = (const float*)d_in[2];
    const float* Wout     = (const float*)d_in[19];
    const float* bout     = (const float*)d_in[20];
    float* outp = (float*)d_out;

    char* ws = (char*)d_ws;
    size_t off = 0;
    auto carve = [&](size_t bytes) -> char* {
        char* p = ws + off;
        off = (off + bytes + 255) & ~(size_t)255;
        return p;
    };

    __bf16* senX = (__bf16*)carve((size_t)BATCH * LSEQ * EPAD * sizeof(__bf16));
    __bf16* tgtX = (__bf16*)carve((size_t)BATCH * TSEQ * EPAD * sizeof(__bf16));
    __bf16* WihB = (__bf16*)carve((size_t)4 * G4 * EPAD * sizeof(__bf16));
    __bf16* WhhB = (__bf16*)carve((size_t)4 * G4 * HDIM * sizeof(__bf16));
    float*  xW_s = (float*)carve((size_t)2 * BATCH * LSEQ * G4 * sizeof(float));
    float*  xW_t = (float*)carve((size_t)2 * BATCH * TSEQ * G4 * sizeof(float));
    float*  hseq_s = (float*)carve((size_t)BATCH * LSEQ * 2 * HDIM * sizeof(float));
    float*  hseq_t = (float*)carve((size_t)BATCH * TSEQ * 2 * HDIM * sizeof(float));

    // K1: embedding gather (padding_idx=0) + bf16 cast
    embed_kernel<<<BATCH * LSEQ, EPAD, 0, stream>>>(sent_idx, embW, senX, BATCH * LSEQ);
    embed_kernel<<<BATCH * TSEQ, EPAD, 0, stream>>>(tgt_idx, embW, tgtX, BATCH * TSEQ);

    // K2: cast weights to bf16 (Wih padded 300->320)
    for (int l = 0; l < 4; ++l) {
        const float* Wih = (const float*)d_in[3 + 4 * l];
        const float* Whh = (const float*)d_in[4 + 4 * l];
        castw_kernel<<<G4, EPAD, 0, stream>>>(Wih, WihB + (size_t)l * G4 * EPAD, EDIM, EPAD);
        castw_kernel<<<G4, HDIM, 0, stream>>>(Whh, WhhB + (size_t)l * G4 * HDIM, HDIM, HDIM);
    }

    // K3: time-hoisted input projections xW = X @ Wih^T + bih + bhh
    for (int l = 0; l < 4; ++l) {
        const float* bih = (const float*)d_in[5 + 4 * l];
        const float* bhh = (const float*)d_in[6 + 4 * l];
        if (l < 2) {
            xw_gemm_kernel<<<dim3(BATCH * LSEQ / 16, 2), 256, 0, stream>>>(
                senX, WihB + (size_t)l * G4 * EPAD, bih, bhh,
                xW_s + (size_t)l * BATCH * LSEQ * G4, BATCH * LSEQ);
        } else {
            xw_gemm_kernel<<<dim3(BATCH * TSEQ / 16, 2), 256, 0, stream>>>(
                tgtX, WihB + (size_t)l * G4 * EPAD, bih, bhh,
                xW_t + (size_t)(l - 2) * BATCH * TSEQ * G4, BATCH * TSEQ);
        }
    }

    // K4: four persistent LSTM recurrences (WMMA h@Whh^T per step, h in LDS)
    lstm_kernel<<<4, 1024, 2 * BATCH * HDIM * sizeof(__bf16), stream>>>(
        WhhB, xW_s, xW_t, hseq_s, hseq_t);

    // K5: attention + classifier
    attn_kernel<<<BATCH, 256, 0, stream>>>(hseq_s, hseq_t, Wout, bout, outp);
}
